// SARoutingBlock_5755256177252
// MI455X (gfx1250) — matrix operand order repaired and checked by reading, not compiled, not run on hardware
//
#include <hip/hip_runtime.h>
#include <hip/hip_bf16.h>

// ---------------------------------------------------------------------------
// SARoutingBlock for MI455X (gfx1250, wave32, WMMA bf16 -> f32 accumulate)
//
// Pipeline:
//   0) transpose_w_kernel x4   : W -> bf16 W^T [N][K] in ws (one-time)
//   1) routing_kernel          : alphas[B,3]   (VALU, tiny)
//   2) gemm_bias<f32->bf16> x3 : q/k/v projections, double-buffered LDS,
//                                async global->LDS B-tile staging
//   3) attn_kernel             : fused QK^T -> 3-order masked softmax
//                                (alpha-combined) -> P*V, async K staging
//   4) gemm_bias<bf16->f32>    : output projection + bias -> d_out (fully async)
// ---------------------------------------------------------------------------

#define HIDDEN 768
#define HEADS  12
#define DK     64
#define SEQ    512
#define BATCH  8
#define ORDERS 3
#define NEGBIG (-1e9f)

typedef __bf16 v16bf __attribute__((ext_vector_type(16)));
typedef float  v8f   __attribute__((ext_vector_type(8)));

// ---- bf16 <-> f32 helpers (RNE) -------------------------------------------
__device__ __forceinline__ __bf16 f2bf(float f) {
    unsigned u = __builtin_bit_cast(unsigned, f);
    unsigned r = (u + 0x7FFFu + ((u >> 16) & 1u)) >> 16;
    unsigned short s = (unsigned short)r;
    return __builtin_bit_cast(__bf16, s);
}
__device__ __forceinline__ float bf2f(__bf16 h) {
    unsigned short s = __builtin_bit_cast(unsigned short, h);
    unsigned u = ((unsigned)s) << 16;
    return __builtin_bit_cast(float, u);
}

// ---- CDNA5 async global->LDS (ASYNCcnt-tracked, ISA cdna5 ch.10 / 15.18) --
// LDS byte address = low 32 bits of the generic pointer (aperture rule:
// LDS offset taken from the low 32 address bits); hardware adds the wave LDS base.
__device__ __forceinline__ void async_load16(void* lds_ptr, const void* gptr) {
    unsigned lds = (unsigned)(uintptr_t)lds_ptr;
    unsigned long long ga = (unsigned long long)(uintptr_t)gptr;
    asm volatile("global_load_async_to_lds_b128 %0, %1, off"
                 :: "v"(lds), "v"(ga)
                 : "memory");
}
__device__ __forceinline__ void wait_async0() {
    asm volatile("s_wait_asynccnt 0" ::: "memory");
}

// ---- WMMA fragment loaders (CDNA5 ISA 7.12.2 layouts, wave32) -------------
// A (16x32, 16-bit): lane L -> M = L&15, half = L>>4.
//   VGPR g, sub r: K = (g<4 ? 2g : 16+2(g-4)) + 8*half + r   (pairs contiguous)
__device__ __forceinline__ v16bf load_frag_a(const __bf16* base, int pitch,
                                             int m0, int k0) {
    int lane = threadIdx.x & 31;
    int m = m0 + (lane & 15);
    int half = lane >> 4;
    v16bf a;
#pragma unroll
    for (int g = 0; g < 8; ++g) {
        int k = k0 + ((g < 4) ? (2 * g) : (16 + 2 * (g - 4))) + 8 * half;
        a[2 * g]     = base[m * pitch + k];
        a[2 * g + 1] = base[m * pitch + k + 1];
    }
    return a;
}
// B (32x16, 16-bit), staged transposed in LDS as [N][K]:
//   lane L -> N = L&15, half = L>>4; VGPR g, sub r: K = (g<4 ? 2g : 16+2(g-4)) + 8*half + r
__device__ __forceinline__ v16bf load_frag_b(const __bf16* base, int pitch,
                                             int n0, int k0) {
    int lane = threadIdx.x & 31;
    int n = n0 + (lane & 15);
    int half = lane >> 4;
    v16bf b;
#pragma unroll
    for (int g = 0; g < 8; ++g) {
        int k = k0 + ((g < 4) ? (2 * g) : (16 + 2 * (g - 4))) + 8 * half;
        b[2 * g]     = base[n * pitch + k];
        b[2 * g + 1] = base[n * pitch + k + 1];
    }
    return b;
}
__device__ __forceinline__ v8f wmma_bf16(v16bf a, v16bf b, v8f c) {
    return __builtin_amdgcn_wmma_f32_16x16x32_bf16(
        false, a, false, b, (short)0, c, false, false);
}

// ---------------------------------------------------------------------------
// 0) One-time weight transpose + f32->bf16: Wt[n*K + k] = bf16(W[k*N + n])
// ---------------------------------------------------------------------------
__global__ __launch_bounds__(256) void transpose_w_kernel(
    const float* __restrict__ W, __bf16* __restrict__ Wt) {
    __shared__ float tile[32][33];
    int bx = blockIdx.x * 32;  // n block
    int by = blockIdx.y * 32;  // k block
    int tx = threadIdx.x & 31, ty = threadIdx.x >> 5;  // ty: 0..7
#pragma unroll
    for (int i = 0; i < 4; ++i) {
        int kr = by + ty + i * 8;
        tile[ty + i * 8][tx] = W[(size_t)kr * HIDDEN + bx + tx];
    }
    __syncthreads();
#pragma unroll
    for (int i = 0; i < 4; ++i) {
        int nr = bx + ty + i * 8;
        Wt[(size_t)nr * HIDDEN + by + tx] = f2bf(tile[tx][ty + i * 8]);
    }
}

// ---------------------------------------------------------------------------
// 1) Soft routing: alphas[B, 3]
// ---------------------------------------------------------------------------
__global__ __launch_bounds__(256) void routing_kernel(
    const float* __restrict__ v, const float* __restrict__ pool_w,
    const float* __restrict__ pool_b, const float* __restrict__ w1,
    const float* __restrict__ w2, const float* __restrict__ b2,
    float* __restrict__ alphas) {
    __shared__ float ls[SEQ];
    __shared__ float red[256];
    __shared__ float lpool[HIDDEN];
    __shared__ float lhid[HIDDEN / 2];
    __shared__ float llog[ORDERS];
    int b = blockIdx.x;
    int tid = threadIdx.x;

    for (int n = tid; n < SEQ; n += 256) {
        const float* x = v + (size_t)(b * SEQ + n) * HIDDEN;
        float sa = 0.f, dot = 0.f;
        for (int h = 0; h < HIDDEN; ++h) {
            float xv = x[h];
            sa += fabsf(xv);
            dot += xv * pool_w[h];
        }
        float sc = dot + pool_b[0];
        if (sa == 0.f) sc = NEGBIG;
        ls[n] = sc;
    }
    __syncthreads();
    float m = fmaxf(ls[tid], ls[tid + 256]);
    red[tid] = m;
    __syncthreads();
    for (int o = 128; o > 0; o >>= 1) {
        if (tid < o) red[tid] = fmaxf(red[tid], red[tid + o]);
        __syncthreads();
    }
    float mx = red[0];
    __syncthreads();
    float e0 = __expf(ls[tid] - mx), e1 = __expf(ls[tid + 256] - mx);
    red[tid] = e0 + e1;
    __syncthreads();
    for (int o = 128; o > 0; o >>= 1) {
        if (tid < o) red[tid] += red[tid + o];
        __syncthreads();
    }
    float inv = 1.f / red[0];
    __syncthreads();
    ls[tid] = e0 * inv;
    ls[tid + 256] = e1 * inv;
    __syncthreads();
    for (int h = tid; h < HIDDEN; h += 256) {
        float acc = 0.f;
        for (int n = 0; n < SEQ; ++n)
            acc += v[(size_t)(b * SEQ + n) * HIDDEN + h] * ls[n];
        lpool[h] = acc;
    }
    __syncthreads();
    for (int j = tid; j < HIDDEN / 2; j += 256) {
        float acc = 0.f;
        for (int h = 0; h < HIDDEN; ++h) acc += lpool[h] * w1[h * (HIDDEN / 2) + j];
        lhid[j] = fmaxf(acc, 0.f);
    }
    __syncthreads();
    if (tid < ORDERS) {
        float acc = b2[tid];
        for (int j = 0; j < HIDDEN / 2; ++j) acc += lhid[j] * w2[j * ORDERS + tid];
        llog[tid] = acc;
    }
    __syncthreads();
    if (tid == 0) {
        float m3 = fmaxf(llog[0], fmaxf(llog[1], llog[2]));
        float e[ORDERS], s = 0.f;
        for (int l = 0; l < ORDERS; ++l) { e[l] = __expf(llog[l] - m3); s += e[l]; }
        for (int l = 0; l < ORDERS; ++l) alphas[b * ORDERS + l] = e[l] / s;
    }
}

// ---------------------------------------------------------------------------
// 2/4) GEMM + bias:  Out[M,N] = A[M,K] * Wt^T + bias   (Wt is bf16 [N][K])
//      block tile 128x128, 8 waves of 32x64, double-buffered LDS,
//      async global->LDS for B (and A when bf16).
// ---------------------------------------------------------------------------
template <bool A_BF16, bool OUT_BF16>
__global__ __launch_bounds__(256) void gemm_bias(
    const void* __restrict__ Ain, const __bf16* __restrict__ Wt,
    const float* __restrict__ bias, void* __restrict__ Out,
    int M, int N, int K) {
    constexpr int KP = 40;  // padded K pitch: 80B rows -> 16B-aligned chunks
    __shared__ __bf16 sA[2][128 * KP];
    __shared__ __bf16 sB[2][128 * KP];  // transposed: [n][k]
    int tid = threadIdx.x;
    int lane = tid & 31, w = tid >> 5;
    int bm0 = blockIdx.x * 128, bn0 = blockIdx.y * 128;
    int wm = (w >> 1) * 32, wn = (w & 1) * 64;

    v8f acc[2][4];
#pragma unroll
    for (int i = 0; i < 2; ++i)
#pragma unroll
        for (int j = 0; j < 4; ++j)
            acc[i][j] = (v8f){0.f, 0.f, 0.f, 0.f, 0.f, 0.f, 0.f, 0.f};

    float ra[16];  // A register staging (f32 path only)

    auto issueB = [&](int kk, int buf) {
#pragma unroll
        for (int i = 0; i < 2; ++i) {
            int idx = tid + i * 256;             // 128 n x 4 chunks
            int n = idx >> 2, c0 = (idx & 3) * 8;
            async_load16(&sB[buf][n * KP + c0],
                         &Wt[(size_t)(bn0 + n) * K + kk + c0]);
        }
    };
    auto issueA = [&](int kk, int buf) {
#pragma unroll
        for (int i = 0; i < 2; ++i) {
            int idx = tid + i * 256;             // 128 r x 4 chunks
            int r = idx >> 2, c0 = (idx & 3) * 8;
            async_load16(&sA[buf][r * KP + c0],
                         &((const __bf16*)Ain)[(size_t)(bm0 + r) * K + kk + c0]);
        }
    };
    auto fetchA = [&](int kk) {
        int r = tid >> 1, c0 = (tid & 1) * 16;
#pragma unroll
        for (int j = 0; j < 16; ++j)
            ra[j] = ((const float*)Ain)[(size_t)(bm0 + r) * K + kk + c0 + j];
    };
    auto storeA = [&](int buf) {
        int r = tid >> 1, c0 = (tid & 1) * 16;
#pragma unroll
        for (int j = 0; j < 16; ++j) sA[buf][r * KP + c0 + j] = f2bf(ra[j]);
    };

    // prologue: tile 0
    issueB(0, 0);
    if constexpr (A_BF16) issueA(0, 0);
    else { fetchA(0); storeA(0); }
    wait_async0();
    __syncthreads();

    int T = K / 32;
    for (int t = 0; t < T; ++t) {
        int cur = t & 1, nxt = cur ^ 1;
        if (t + 1 < T) {  // overlap next tile's loads with this tile's WMMAs
            issueB((t + 1) * 32, nxt);
            if constexpr (A_BF16) issueA((t + 1) * 32, nxt);
            else fetchA((t + 1) * 32);
        }
        v16bf a0 = load_frag_a(sA[cur], KP, wm + 0, 0);
        v16bf a1 = load_frag_a(sA[cur], KP, wm + 16, 0);
#pragma unroll
        for (int nt = 0; nt < 4; ++nt) {
            v16bf bb = load_frag_b(sB[cur], KP, wn + nt * 16, 0);
            acc[0][nt] = wmma_bf16(a0, bb, acc[0][nt]);
            acc[1][nt] = wmma_bf16(a1, bb, acc[1][nt]);
        }
        if (t + 1 < T) {
            if constexpr (!A_BF16) storeA(nxt);
            wait_async0();     // own async loads complete
            __syncthreads();   // all waves' loads/stores visible
        }
    }
    // epilogue (C layout: M = i + 8*(lane>=16), N = lane&15)
    int halfc = lane >> 4, nloc = lane & 15;
#pragma unroll
    for (int mt = 0; mt < 2; ++mt)
#pragma unroll
        for (int nt = 0; nt < 4; ++nt)
#pragma unroll
            for (int i = 0; i < 8; ++i) {
                int row = bm0 + wm + mt * 16 + i + 8 * halfc;
                int col = bn0 + wn + nt * 16 + nloc;
                float vo = acc[mt][nt][i] + bias[col];
                if constexpr (OUT_BF16)
                    ((__bf16*)Out)[(size_t)row * N + col] = f2bf(vo);
                else
                    ((float*)Out)[(size_t)row * N + col] = vo;
            }
}

// ---------------------------------------------------------------------------
// 3) Fused attention per (b, h, 32-row tile):
//    S = (Q Kt)/8 in LDS -> 3-order masked softmax, alpha-mix -> P (bf16)
//    O = P * V  -> atted (bf16, [B*N, HIDDEN] layout)
// ---------------------------------------------------------------------------
__global__ __launch_bounds__(256) void attn_kernel(
    const __bf16* __restrict__ qp, const __bf16* __restrict__ kp,
    const __bf16* __restrict__ vp, const unsigned char* __restrict__ masks,
    const float* __restrict__ alphas, __bf16* __restrict__ atted) {
    constexpr int SP = 520;  // score row pitch (f32)
    constexpr int QP = 72;   // Q/K tile pitch: 144B rows -> 16B-aligned chunks
    __shared__ float  sS[32 * SP];     //  66.5 KB
    __shared__ __bf16 sP[32 * SP];     //  33.3 KB
    __shared__ __bf16 sQ[32 * QP];     //   4.6 KB
    __shared__ __bf16 sKV[2][64 * QP]; //  18.4 KB (double-buffered K / V^T)

    int tid = threadIdx.x;
    int lane = tid & 31, w = tid >> 5;
    int rt = blockIdx.x;               // 16 row tiles of 32
    int bh = blockIdx.y;               // B*H
    int b = bh / HEADS, h = bh % HEADS;
    int row0 = rt * 32;

    auto issueK = [&](int kt, int buf) {
#pragma unroll
        for (int i = 0; i < 2; ++i) {
            int idx = tid + i * 256;           // 64 keys x 8 chunks
            int n = idx >> 3, c0 = (idx & 7) * 8;
            async_load16(&sKV[buf][n * QP + c0],
                         &kp[(size_t)(b * SEQ + kt * 64 + n) * HIDDEN + h * DK + c0]);
        }
    };
    // stage Q tile 32x64 async (one 16B chunk per thread)
    {
        int r = tid >> 3, c0 = (tid & 7) * 8;
        async_load16(&sQ[r * QP + c0],
                     &qp[(size_t)(b * SEQ + row0 + r) * HIDDEN + h * DK + c0]);
    }
    issueK(0, 0);
    wait_async0();
    __syncthreads();

    // ---- phase 1: scores S[32 x 512] = Q Kt / sqrt(dk) --------------------
    {
        int mt = w >> 2, nt = w & 3;   // 2 x 4 tiles per 64-key slab
        for (int kt = 0; kt < 8; ++kt) {
            int cur = kt & 1;
            if (kt + 1 < 8) issueK(kt + 1, cur ^ 1);  // overlap with WMMAs
            v8f acc = (v8f){0.f, 0.f, 0.f, 0.f, 0.f, 0.f, 0.f, 0.f};
#pragma unroll
            for (int ks = 0; ks < 2; ++ks) {
                v16bf a  = load_frag_a(sQ, QP, mt * 16, ks * 32);
                v16bf bb = load_frag_b(sKV[cur], QP, nt * 16, ks * 32);
                acc = wmma_bf16(a, bb, acc);
            }
            int halfc = lane >> 4, nloc = lane & 15;
#pragma unroll
            for (int i = 0; i < 8; ++i)
                sS[(mt * 16 + i + 8 * halfc) * SP + kt * 64 + nt * 16 + nloc] =
                    acc[i] * 0.125f;
            wait_async0();
            __syncthreads();
        }
    }

    // ---- phase 2: per-row 3-order masked softmax, combined by alphas ------
    float al[ORDERS];
#pragma unroll
    for (int l = 0; l < ORDERS; ++l) al[l] = alphas[b * ORDERS + l];
    for (int rr = w * 4; rr < w * 4 + 4; ++rr) {
        int grow = row0 + rr;
        int c0 = lane * 16;
        float s[16];
#pragma unroll
        for (int j = 0; j < 16; ++j) s[j] = sS[rr * SP + c0 + j];
        float pacc[16];
#pragma unroll
        for (int j = 0; j < 16; ++j) pacc[j] = 0.f;
#pragma unroll
        for (int l = 0; l < ORDERS; ++l) {
            const unsigned char* mrow =
                masks + ((size_t)(l * BATCH + b) * SEQ + grow) * SEQ + c0;
            unsigned bits = 0;
#pragma unroll
            for (int j = 0; j < 16; ++j) bits |= (mrow[j] ? 1u : 0u) << j;
            float mx = -3.4e38f;
#pragma unroll
            for (int j = 0; j < 16; ++j) {
                float t = ((bits >> j) & 1u) ? NEGBIG : s[j];
                mx = fmaxf(mx, t);
            }
#pragma unroll
            for (int o = 16; o >= 1; o >>= 1) mx = fmaxf(mx, __shfl_xor(mx, o, 32));
            float e[16], se = 0.f;
#pragma unroll
            for (int j = 0; j < 16; ++j) {
                e[j] = ((bits >> j) & 1u) ? 0.f : __expf(s[j] - mx);
                se += e[j];
            }
#pragma unroll
            for (int o = 16; o >= 1; o >>= 1) se += __shfl_xor(se, o, 32);
            float inv = al[l] / fmaxf(se, 1e-30f);
#pragma unroll
            for (int j = 0; j < 16; ++j) pacc[j] += e[j] * inv;
        }
#pragma unroll
        for (int j = 0; j < 16; ++j) sP[rr * SP + c0 + j] = f2bf(pacc[j]);
    }
    __syncthreads();

    // ---- phase 3: O[32x64] = P[32x512] * V[512x64] ------------------------
    int mt = w >> 2, nt = w & 3;  // 2 x 4 = 8 output tiles, one per wave
    v8f acc = (v8f){0.f, 0.f, 0.f, 0.f, 0.f, 0.f, 0.f, 0.f};
    __bf16* sVT = &sKV[0][0];     // V^T view: [dk 64][key 32], pitch 36
    for (int kt = 0; kt < 16; ++kt) {   // 32-key slabs
#pragma unroll
        for (int i = 0; i < 8; ++i) {
            int idx = tid + i * 256;
            int kl = idx >> 6, d = idx & 63;
            sVT[d * 36 + kl] =
                vp[(size_t)(b * SEQ + kt * 32 + kl) * HIDDEN + h * DK + d];
        }
        __syncthreads();
        v16bf a  = load_frag_a(sP, SP, mt * 16, kt * 32);
        v16bf bb = load_frag_b(sVT, 36, nt * 16, 0);
        acc = wmma_bf16(a, bb, acc);
        __syncthreads();
    }
    int halfc = lane >> 4, nloc = lane & 15;
#pragma unroll
    for (int i = 0; i < 8; ++i) {
        int row = row0 + mt * 16 + i + 8 * halfc;
        int col = h * DK + nt * 16 + nloc;
        atted[(size_t)(b * SEQ + row) * HIDDEN + col] = f2bf(acc[i]);
    }
}

// ---------------------------------------------------------------------------
extern "C" void kernel_launch(void* const* d_in, const int* in_sizes, int n_in,
                              void* d_out, int out_size, void* d_ws,
                              size_t ws_size, hipStream_t stream) {
    const float* v  = (const float*)d_in[0];
    const float* k  = (const float*)d_in[1];
    const float* q  = (const float*)d_in[2];
    const unsigned char* masks = (const unsigned char*)d_in[3];
    // d_in[4]=tau, d_in[5]=training: unused by the reference math
    const float* Wv = (const float*)d_in[6];
    const float* bv = (const float*)d_in[7];
    const float* Wk = (const float*)d_in[8];
    const float* bk = (const float*)d_in[9];
    const float* Wq = (const float*)d_in[10];
    const float* bq = (const float*)d_in[11];
    const float* Wm = (const float*)d_in[12];
    const float* bm = (const float*)d_in[13];
    const float* pool_w = (const float*)d_in[14];
    const float* pool_b = (const float*)d_in[15];
    const float* mlp_w1 = (const float*)d_in[16];
    const float* mlp_w2 = (const float*)d_in[17];
    const float* mlp_b2 = (const float*)d_in[18];
    float* out = (float*)d_out;

    // workspace layout
    char* ws = (char*)d_ws;
    const size_t MAT  = (size_t)BATCH * SEQ * HIDDEN;   // 3,145,728 elems
    const size_t WMAT = (size_t)HIDDEN * HIDDEN;        //   589,824 elems
    float*  alphas = (float*)ws;
    __bf16* qp    = (__bf16*)(ws + 256);
    __bf16* kp    = (__bf16*)(ws + 256 + MAT * 2);
    __bf16* vp    = (__bf16*)(ws + 256 + MAT * 4);
    __bf16* atted = (__bf16*)(ws + 256 + MAT * 6);
    __bf16* Wqt   = (__bf16*)(ws + 256 + MAT * 8);
    __bf16* Wkt   = (__bf16*)(ws + 256 + MAT * 8 + WMAT * 2);
    __bf16* Wvt   = (__bf16*)(ws + 256 + MAT * 8 + WMAT * 4);
    __bf16* Wmt   = (__bf16*)(ws + 256 + MAT * 8 + WMAT * 6);

    const int M = BATCH * SEQ;  // 4096

    dim3 gt(HIDDEN / 32, HIDDEN / 32);  // 24 x 24
    transpose_w_kernel<<<gt, 256, 0, stream>>>(Wq, Wqt);
    transpose_w_kernel<<<gt, 256, 0, stream>>>(Wk, Wkt);
    transpose_w_kernel<<<gt, 256, 0, stream>>>(Wv, Wvt);
    transpose_w_kernel<<<gt, 256, 0, stream>>>(Wm, Wmt);

    routing_kernel<<<BATCH, 256, 0, stream>>>(v, pool_w, pool_b, mlp_w1, mlp_w2,
                                              mlp_b2, alphas);

    dim3 gg(M / 128, HIDDEN / 128);  // 32 x 6
    gemm_bias<false, true><<<gg, 256, 0, stream>>>(q, Wqt, bq, qp, M, HIDDEN, HIDDEN);
    gemm_bias<false, true><<<gg, 256, 0, stream>>>(k, Wkt, bk, kp, M, HIDDEN, HIDDEN);
    gemm_bias<false, true><<<gg, 256, 0, stream>>>(v, Wvt, bv, vp, M, HIDDEN, HIDDEN);

    dim3 ga(SEQ / 32, BATCH * HEADS);  // 16 x 96
    attn_kernel<<<ga, 256, 0, stream>>>(qp, kp, vp, masks, alphas, atted);

    gemm_bias<true, false><<<gg, 256, 0, stream>>>(atted, Wmt, bm, out, M, HIDDEN,
                                                   HIDDEN);
}
